// AugmentedNeuralODE_23871428231355
// MI455X (gfx1250) — matrix operand order repaired and checked by reading, compiled, not run on hardware
//
#include <hip/hip_runtime.h>
#include <hip/hip_bf16.h>

// ---------------------------------------------------------------------------
// Augmented Neural ODE forward, MI455X (gfx1250, wave32, WMMA bf16).
//  - y state kept in f32 (Euler accumulation exact-ish); GEMMs in bf16 with
//    f32 accumulation via v_wmma_f32_16x16x32_bf16.
//  - Big layers: 256x128 block tile, 8 waves of 64x64, LDS double-buffered
//    K-stages (XOR-swizzled, bank-conflict-free fragment reads), staged with
//    GLOBAL_LOAD_ASYNC_TO_LDS_B128 when available (ASYNCcnt path).
// ---------------------------------------------------------------------------

typedef __attribute__((ext_vector_type(16))) __bf16 v16bf;
typedef __attribute__((ext_vector_type(8)))  __bf16 v8bf;
typedef __attribute__((ext_vector_type(8)))  float  v8f;
typedef __attribute__((ext_vector_type(4)))  int    v4i;

#define DEVI static __device__ __forceinline__

#if defined(__gfx1250__) && \
    __has_builtin(__builtin_amdgcn_global_load_async_to_lds_b128)
#define USE_ASYNC_LDS 1
#else
#define USE_ASYNC_LDS 0
#endif

DEVI __bf16 f2bf(float x) { return (__bf16)x; }

DEVI v16bf cat8(v8bf lo, v8bf hi) {
  return __builtin_shufflevector(lo, hi, 0, 1, 2, 3, 4, 5, 6, 7, 8, 9, 10, 11,
                                 12, 13, 14, 15);
}

// ===========================================================================
// LDS-staged GEMM: out = bf16(tanh(A[M,K] * W[N,K]^T + bias)).
// Block tile 256x128, KC=32 per stage, 8 waves (4x2) of 64x64 each.
//
// LDS tile layout: per row (A-row or W-row), 32 bf16 = 64 B = four 16-B quads.
// Quad q of row r lives at byte  r*64 + 16*(q ^ ((r>>2)&3))   (XOR swizzle,
// conflict-free fragment reads).  A is written with quad permutation
// {0,2,1,3} so quads hold k = {0-7, 16-23, 8-15, 24-31}: a lane's 32-B
// fragment (ISA 7.12.2 half-lane K split) is then quads {2h, 2h+1} for both
// A and B.
// ===========================================================================
constexpr int BM = 256, BN = 128, KC = 32;
constexpr int A_TILE_B = BM * 64;  // 16 KB
constexpr int B_TILE_B = BN * 64;  // 8 KB

DEVI int qswz(int row, int q) { return (row << 6) + ((q ^ ((row >> 2) & 3)) << 4); }

DEVI v16bf lds_frag(const char* tile, int row, int half) {
  const int q0 = half * 2;
  v8bf lo = *(const v8bf*)(tile + qswz(row, q0));
  v8bf hi = *(const v8bf*)(tile + qswz(row, q0 + 1));
  return cat8(lo, hi);
}

#if USE_ASYNC_LDS
DEVI void async_copy16(const __bf16* g, char* l) {
  __builtin_amdgcn_global_load_async_to_lds_b128(
      (__attribute__((address_space(1))) v4i*)g,
      (__attribute__((address_space(3))) v4i*)l, 0, 0);
}
DEVI void wait_async0() {
#if __has_builtin(__builtin_amdgcn_s_wait_asynccnt)
  __builtin_amdgcn_s_wait_asynccnt(0);
#else
  asm volatile("s_wait_asynccnt 0" ::: "memory");
#endif
}
#endif

__global__ __launch_bounds__(256) void gemm_lds_tanh(
    const __bf16* __restrict__ A, int lda, const __bf16* __restrict__ W,
    int ldw, const float* __restrict__ bias, int K, __bf16* __restrict__ outB,
    int ldo) {
  __shared__ char sm[2][A_TILE_B + B_TILE_B];  // 48 KB

  const int t = threadIdx.x;
  const int lane = t & 31;
  const int wave = t >> 5;
  const int wrow = wave >> 1;  // 0..3 -> 64-row band
  const int wcol = wave & 1;   // 0..1 -> 64-col band
  const int m0 = blockIdx.x * BM;
  const int n0 = blockIdx.y * BN;
  const int r = lane & 15;
  const int half = lane >> 4;

  // Staging assignment: thread t moves quad (row = t>>2 + i*64, kg = t&3).
  const int srow = t >> 2;  // 0..63
  const int kg = t & 3;
  const int pkg = (kg == 1) ? 2 : (kg == 2 ? 1 : kg);  // A quad permutation

  v8f acc[4][4];
#pragma unroll
  for (int m = 0; m < 4; ++m)
#pragma unroll
    for (int s = 0; s < 4; ++s)
      acc[m][s] = (v8f){0.f, 0.f, 0.f, 0.f, 0.f, 0.f, 0.f, 0.f};

  const int nst = K / KC;

#if USE_ASYNC_LDS
  auto stage = [&](int ks, int buf) {
    char* bufA = &sm[buf][0];
    char* bufB = &sm[buf][A_TILE_B];
#pragma unroll
    for (int i = 0; i < 4; ++i) {
      const int row = i * 64 + srow;
      async_copy16(A + (size_t)(m0 + row) * lda + ks * KC + kg * 8,
                   bufA + qswz(row, pkg));
    }
#pragma unroll
    for (int i = 0; i < 2; ++i) {
      const int row = i * 64 + srow;
      async_copy16(W + (size_t)(n0 + row) * ldw + ks * KC + kg * 8,
                   bufB + qswz(row, pkg));
    }
  };
#else
  v8bf ra[4], rb[2];
  auto fetch = [&](int ks) {
#pragma unroll
    for (int i = 0; i < 4; ++i)
      ra[i] = *(const v8bf*)(A + (size_t)(m0 + i * 64 + srow) * lda + ks * KC +
                             kg * 8);
#pragma unroll
    for (int i = 0; i < 2; ++i)
      rb[i] = *(const v8bf*)(W + (size_t)(n0 + i * 64 + srow) * ldw + ks * KC +
                             kg * 8);
  };
  auto store = [&](int buf) {
    char* bufA = &sm[buf][0];
    char* bufB = &sm[buf][A_TILE_B];
#pragma unroll
    for (int i = 0; i < 4; ++i)
      *(v8bf*)(bufA + qswz(i * 64 + srow, pkg)) = ra[i];
#pragma unroll
    for (int i = 0; i < 2; ++i)
      *(v8bf*)(bufB + qswz(i * 64 + srow, pkg)) = rb[i];
  };
#endif

  auto compute = [&](int buf) {
    const char* bufA = &sm[buf][0];
    const char* bufB = &sm[buf][A_TILE_B];
    v16bf bfr[4];
#pragma unroll
    for (int s = 0; s < 4; ++s)
      bfr[s] = lds_frag(bufB, wcol * 64 + s * 16 + r, half);
#pragma unroll
    for (int m = 0; m < 4; ++m) {
      v16bf af = lds_frag(bufA, wrow * 64 + m * 16 + r, half);
#pragma unroll
      for (int s = 0; s < 4; ++s)
        acc[m][s] = __builtin_amdgcn_wmma_f32_16x16x32_bf16(
            false, af, false, bfr[s], (short)0, acc[m][s], false, false);
    }
  };

#if USE_ASYNC_LDS
  stage(0, 0);
  wait_async0();
  __syncthreads();
  for (int ks = 0; ks < nst; ++ks) {
    const int cur = ks & 1;
    if (ks + 1 < nst) stage(ks + 1, cur ^ 1);
    compute(cur);
    wait_async0();
    __syncthreads();
  }
#else
  fetch(0);
  store(0);
  __syncthreads();
  for (int ks = 0; ks < nst; ++ks) {
    const int cur = ks & 1;
    if (ks + 1 < nst) fetch(ks + 1);
    compute(cur);
    if (ks + 1 < nst) store(cur ^ 1);
    __syncthreads();
  }
#endif

  // Epilogue: C/D layout -> col = lane%16 (+16*s), row = v + 8*half (+16*m).
#pragma unroll
  for (int m = 0; m < 4; ++m) {
#pragma unroll
    for (int s = 0; s < 4; ++s) {
      const int col = n0 + wcol * 64 + s * 16 + r;
      const float bv = bias[col];
#pragma unroll
      for (int v = 0; v < 8; ++v) {
        const int row = m0 + wrow * 64 + m * 16 + v + 8 * half;
        outB[(size_t)row * ldo + col] = f2bf(tanhf(acc[m][s][v] + bv));
      }
    }
  }
}

// ===========================================================================
// Direct-load GEMM (small layers): augmentation + Euler update epilogues.
// ===========================================================================
DEVI v16bf load_a_frag(const __bf16* p) {
  return cat8(*(const v8bf*)(p), *(const v8bf*)(p + 16));
}
DEVI v16bf load_b_frag(const __bf16* p) { return *(const v16bf*)(p); }

enum { EPI_AUG = 1, EPI_EULER = 2 };

template <int TBM, int TBN, int MODE>
__global__ __launch_bounds__(256) void gemm_bf16_wmma(
    const __bf16* __restrict__ A, int lda, const __bf16* __restrict__ W,
    int ldw, const float* __restrict__ bias, int K, float* __restrict__ yF,
    __bf16* __restrict__ yB, const float* __restrict__ tvec) {
  constexpr int WCOLS = TBN / 32;
  constexpr int WROWS = 8 / WCOLS;
  constexpr int WM = TBM / WROWS;
  constexpr int MS = WM / 16;

  const int wave = threadIdx.x >> 5;
  const int lane = threadIdx.x & 31;
  const int wrow = wave / WCOLS;
  const int wcol = wave % WCOLS;
  const int mbase = blockIdx.x * TBM + wrow * WM;
  const int nbase = blockIdx.y * TBN + wcol * 32;
  const int r = lane & 15;
  const int half = lane >> 4;

  v8f acc[MS][2];
#pragma unroll
  for (int m = 0; m < MS; ++m)
#pragma unroll
    for (int s = 0; s < 2; ++s)
      acc[m][s] = (v8f){0.f, 0.f, 0.f, 0.f, 0.f, 0.f, 0.f, 0.f};

  for (int k = 0; k < K; k += 32) {
    v16bf bfr[2];
#pragma unroll
    for (int s = 0; s < 2; ++s)
      bfr[s] =
          load_a_frag(W + (size_t)(nbase + s * 16 + r) * ldw + k + half * 8);
#pragma unroll
    for (int m = 0; m < MS; ++m) {
      v16bf af =
          load_a_frag(A + (size_t)(mbase + m * 16 + r) * lda + k + half * 8);
#pragma unroll
      for (int s = 0; s < 2; ++s)
        acc[m][s] = __builtin_amdgcn_wmma_f32_16x16x32_bf16(
            false, af, false, bfr[s], (short)0, acc[m][s], false, false);
    }
  }

  float dt = 0.f;
  if (MODE == EPI_EULER) dt = tvec[1] - tvec[0];

#pragma unroll
  for (int m = 0; m < MS; ++m) {
#pragma unroll
    for (int s = 0; s < 2; ++s) {
      const int col = nbase + s * 16 + r;
      const float bv = bias[col];
#pragma unroll
      for (int v = 0; v < 8; ++v) {
        const int row = mbase + m * 16 + v + half * 8;
        const float x = acc[m][s][v] + bv;
        if constexpr (MODE == EPI_AUG) {
          yF[(size_t)row * 128 + 64 + col] = x;
          yB[(size_t)row * 128 + 64 + col] = f2bf(x);
        } else {
          const float yn = yF[(size_t)row * 128 + col] + dt * x;
          yF[(size_t)row * 128 + col] = yn;
          yB[(size_t)row * 128 + col] = f2bf(yn);
        }
      }
    }
  }
}

// ===========================================================================
// Small helpers.
// ===========================================================================
__global__ void cvt_f32_bf16(const float* __restrict__ in,
                             __bf16* __restrict__ out, int n) {
  int i = blockIdx.x * blockDim.x + threadIdx.x;
  if (i < n) out[i] = f2bf(in[i]);
}

__global__ void init_state(const float* __restrict__ y0,
                           float* __restrict__ yF, __bf16* __restrict__ yB,
                           int n) {
  int i = blockIdx.x * blockDim.x + threadIdx.x;
  if (i < n) {
    int row = i >> 6, c = i & 63;
    float v = y0[i];
    yF[(size_t)row * 128 + c] = v;
    yB[(size_t)row * 128 + c] = f2bf(v);
  }
}

__global__ void extract_out(const float* __restrict__ yF,
                            float* __restrict__ out, int n) {
  int i = blockIdx.x * blockDim.x + threadIdx.x;
  if (i < n) out[i] = yF[(size_t)(i >> 6) * 128 + (i & 63)];
}

// ===========================================================================
extern "C" void kernel_launch(void* const* d_in, const int* in_sizes, int n_in,
                              void* d_out, int out_size, void* d_ws,
                              size_t ws_size, hipStream_t stream) {
  (void)in_sizes; (void)n_in; (void)out_size; (void)ws_size;

  constexpr int M = 64 * 64;  // B*S rows
  constexpr int DIN = 64, STATE = 128, HID = 1024, T = 32;

  const float* y0 = (const float*)d_in[0];
  const float* t = (const float*)d_in[1];
  const float* W_aug = (const float*)d_in[2];
  const float* b_aug = (const float*)d_in[3];
  const float* W0 = (const float*)d_in[4];
  const float* b0 = (const float*)d_in[5];
  const float* W1 = (const float*)d_in[6];
  const float* b1 = (const float*)d_in[7];
  const float* W2 = (const float*)d_in[8];
  const float* b2 = (const float*)d_in[9];
  const float* W3 = (const float*)d_in[10];
  const float* b3 = (const float*)d_in[11];

  char* base = (char*)d_ws;
  size_t off = 0;
  auto alloc = [&](size_t bytes) {
    char* p = base + off;
    off += (bytes + 255) & ~(size_t)255;
    return p;
  };
  float* yF = (float*)alloc((size_t)M * STATE * 4);
  __bf16* yB = (__bf16*)alloc((size_t)M * STATE * 2);
  __bf16* hA = (__bf16*)alloc((size_t)M * HID * 2);
  __bf16* hB = (__bf16*)alloc((size_t)M * HID * 2);
  __bf16* wAugB = (__bf16*)alloc((size_t)DIN * DIN * 2);
  __bf16* w0B = (__bf16*)alloc((size_t)HID * STATE * 2);
  __bf16* w1B = (__bf16*)alloc((size_t)HID * HID * 2);
  __bf16* w2B = (__bf16*)alloc((size_t)HID * HID * 2);
  __bf16* w3B = (__bf16*)alloc((size_t)STATE * HID * 2);

  auto cvt = [&](const float* src, __bf16* dst, int n) {
    cvt_f32_bf16<<<(n + 255) / 256, 256, 0, stream>>>(src, dst, n);
  };
  cvt(W_aug, wAugB, DIN * DIN);
  cvt(W0, w0B, HID * STATE);
  cvt(W1, w1B, HID * HID);
  cvt(W2, w2B, HID * HID);
  cvt(W3, w3B, STATE * HID);

  init_state<<<(M * DIN + 255) / 256, 256, 0, stream>>>(y0, yF, yB, M * DIN);

  // Augmentation: y[:,64:128] = y0 @ W_aug^T + b_aug   (M x 64, K = 64)
  gemm_bf16_wmma<128, 64, EPI_AUG><<<dim3(M / 128, 1), 256, 0, stream>>>(
      yB, STATE, wAugB, DIN, b_aug, DIN, yF, yB, nullptr);

  for (int step = 0; step < T - 1; ++step) {
    // h1 = tanh(y @ W0^T + b0)   [M,1024], K=128
    gemm_lds_tanh<<<dim3(M / BM, HID / BN), 256, 0, stream>>>(
        yB, STATE, w0B, STATE, b0, STATE, hA, HID);
    // h2 = tanh(h1 @ W1^T + b1)  [M,1024], K=1024
    gemm_lds_tanh<<<dim3(M / BM, HID / BN), 256, 0, stream>>>(
        hA, HID, w1B, HID, b1, HID, hB, HID);
    // h3 = tanh(h2 @ W2^T + b2)  [M,1024], K=1024
    gemm_lds_tanh<<<dim3(M / BM, HID / BN), 256, 0, stream>>>(
        hB, HID, w2B, HID, b2, HID, hA, HID);
    // y += dt * (h3 @ W3^T + b3)  [M,128], K=1024
    gemm_bf16_wmma<64, 128, EPI_EULER><<<dim3(M / 64, 1), 256, 0, stream>>>(
        hA, HID, w3B, HID, b3, HID, yF, yB, t);
  }

  extract_out<<<(M * DIN + 255) / 256, 256, 0, stream>>>(yF, (float*)d_out,
                                                         M * DIN);
}
